// Attention_8624294330869
// MI455X (gfx1250) — compile-verified, matrix operand order checked by reading, not run on hardware
//
#include <hip/hip_runtime.h>

typedef __attribute__((ext_vector_type(16))) __bf16 v16bf;
typedef __attribute__((ext_vector_type(8)))  float  v8f;
typedef unsigned int u32x4 __attribute__((ext_vector_type(4)));
typedef int          i32x4 __attribute__((ext_vector_type(4)));
typedef int          i32x8 __attribute__((ext_vector_type(8)));

#define WMMA_BF16(A, B, C) \
    __builtin_amdgcn_wmma_f32_16x16x32_bf16(false, (A), false, (B), (short)0, (C), false, false)

#if defined(__has_builtin)
#if __has_builtin(__builtin_amdgcn_tensor_load_to_lds) && \
    __has_builtin(__builtin_amdgcn_s_wait_tensorcnt)
#define USE_TDM 1
#endif
#endif

// ---------------------------------------------------------------- helpers ---

static __device__ __forceinline__ unsigned short f2bf(float f) {
    unsigned u = __float_as_uint(f);
    u += 0x7FFFu + ((u >> 16) & 1u);   // round-to-nearest-even
    return (unsigned short)(u >> 16);
}

union FragU  { uint4 u[2]; v16bf v; };
union Chunk8 { uint4 u; unsigned short s[8]; };

// Load one 16x32 bf16 WMMA fragment: two contiguous 16B chunks per lane.
static __device__ __forceinline__ v16bf ld_frag(const unsigned short* p, int o0, int o1) {
    FragU f;
    f.u[0] = *(const uint4*)(p + o0);
    f.u[1] = *(const uint4*)(p + o1);
    return f.v;
}

static __device__ __forceinline__ v8f vzero() {
    v8f z = {0.f, 0.f, 0.f, 0.f, 0.f, 0.f, 0.f, 0.f};
    return z;
}

// ------------------------------------------------------------- convert ------

__global__ void cvt_bf16(const float* __restrict__ src, unsigned short* __restrict__ dst, int n) {
    int i = blockIdx.x * blockDim.x + threadIdx.x;
    if (i < n) dst[i] = f2bf(src[i]);
}

// ------------------------------------------------------------- qkv gemm -----
// C[8192][3072] = x_bf[8192][1024] * w_bf[3072][1024]^T, scattered into
// q/k/v[bh=64][n=2048][d=64] (q pre-scaled by 0.125).

__global__ __launch_bounds__(256) void qkv_gemm(
    const unsigned short* __restrict__ xb,
    const unsigned short* __restrict__ wb,
    unsigned short* __restrict__ qd,
    unsigned short* __restrict__ kd,
    unsigned short* __restrict__ vd)
{
    __shared__ unsigned short As[128 * 40];
    __shared__ unsigned short Bs[128 * 40];

    const int tid  = threadIdx.x;
    const int m0   = blockIdx.y * 128;
    const int n0   = blockIdx.x * 128;
    const int wv   = tid >> 5;
    const int lane = tid & 31;
    const int wr   = (wv >> 2) * 64;
    const int wc   = (wv & 3) * 32;
    const int lr   = lane & 15;
    const int koff = (lane >> 4) * 8;

    v8f acc[4][2];
#pragma unroll
    for (int i = 0; i < 4; ++i)
#pragma unroll
        for (int j = 0; j < 2; ++j) acc[i][j] = vzero();

    for (int k0 = 0; k0 < 1024; k0 += 32) {
#pragma unroll
        for (int i = 0; i < 2; ++i) {
            int c   = tid + i * 256;
            int row = c >> 2;
            int k8  = (c & 3) * 8;
            *(uint4*)&As[row * 40 + k8] = *(const uint4*)(xb + (size_t)(m0 + row) * 1024 + k0 + k8);
            *(uint4*)&Bs[row * 40 + k8] = *(const uint4*)(wb + (size_t)(n0 + row) * 1024 + k0 + k8);
            if (k0 + 32 < 1024) {
                __builtin_prefetch(xb + (size_t)(m0 + row) * 1024 + k0 + 32 + k8, 0, 0);
                __builtin_prefetch(wb + (size_t)(n0 + row) * 1024 + k0 + 32 + k8, 0, 0);
            }
        }
        __syncthreads();

        v16bf a[4], b[2];
#pragma unroll
        for (int i = 0; i < 4; ++i)
            a[i] = ld_frag(&As[(wr + i * 16 + lr) * 40], koff, koff + 16);
#pragma unroll
        for (int j = 0; j < 2; ++j)
            b[j] = ld_frag(&Bs[(wc + j * 16 + lr) * 40], koff, koff + 16);

#pragma unroll
        for (int i = 0; i < 4; ++i)
#pragma unroll
            for (int j = 0; j < 2; ++j)
                acc[i][j] = WMMA_BF16(a[i], b[j], acc[i][j]);
        __syncthreads();
    }

#pragma unroll
    for (int i = 0; i < 4; ++i) {
        int rbase = m0 + wr + i * 16 + (lane >> 4) * 8;
#pragma unroll
        for (int j = 0; j < 2; ++j) {
            int col   = n0 + wc + j * 16 + lr;
            int three = col >> 10;
            int h     = (col >> 6) & 15;
            int d     = col & 63;
            unsigned short* dst = (three == 0) ? qd : ((three == 1) ? kd : vd);
            float sc = (three == 0) ? 0.125f : 1.0f;   // SCALE = 64^-0.5
#pragma unroll
            for (int r = 0; r < 8; ++r) {
                int row = rbase + r;
                int bb  = row >> 11;
                int nr  = row & 2047;
                size_t idx = (((size_t)(bb * 16 + h)) * 2048 + nr) * 64 + d;
                dst[idx] = f2bf(acc[i][j][r] * sc);
            }
        }
    }
}

// ---------------------------------------------------------- flash attn ------
// Grid: x = 32 q-blocks of 64, y = 64 (b*h). Block 128 thr = 4 waves;
// each wave owns 16 q rows, full d=64. Online softmax, 16 WMMA per k-block.
// K tile DMA'd into LDS by the Tensor Data Mover (wave 0), with TDM padding
// producing the 72-halfword bank-conflict-free row stride.

__global__ __launch_bounds__(128) void flash_attn(
    const unsigned short* __restrict__ q,
    const unsigned short* __restrict__ k,
    const unsigned short* __restrict__ v,
    unsigned short* __restrict__ ob)       // [4][2048][1024] bf16
{
    __shared__ unsigned short Ks[64 * 72];      // [krow][d], pad 72
    __shared__ unsigned short Vt[64 * 72];      // [d][krow] (transposed), pad 72
    __shared__ unsigned short Ps[4][16 * 72];   // wave-private P tiles

    const int tid  = threadIdx.x;
    const int wv   = tid >> 5;
    const int lane = tid & 31;
    const int lr   = lane & 15;
    const int koff = (lane >> 4) * 8;
    const int bh   = blockIdx.y;
    const int q0   = blockIdx.x * 64 + wv * 16;

    const unsigned short* qp = q + ((size_t)bh * 2048 + q0 + lr) * 64;
    const v16bf qa0 = ld_frag(qp, koff, koff + 16);
    const v16bf qa1 = ld_frag(qp, 32 + koff, 32 + koff + 16);

    float m[8], l[8];
    v8f o[4];
#pragma unroll
    for (int r = 0; r < 8; ++r) { m[r] = -1e30f; l[r] = 0.f; }
#pragma unroll
    for (int t = 0; t < 4; ++t) o[t] = vzero();

    unsigned short* pw = Ps[wv];

#ifdef USE_TDM
    // D# group1 (constant): data_size=2B, pad every 32 DWORDs by 4 DWORDs,
    // 1-D tensor: tensor_dim0 = tile_dim0 = 4096 elems, stride 4096.
    const unsigned lds_ks = (unsigned)(unsigned long long)(const void*)&Ks[0];
    i32x8 g1;
    g1[0] = (1 << 16) | (1 << 20) | (4 << 22) | (3 << 25);
    g1[1] = (int)(4096u << 16);     // tensor_dim0[15:0] in bits[63:48]
    g1[2] = (1 << 16);              // tensor_dim0 hi = 0, tensor_dim1 = 1
    g1[3] = (int)(4096u << 16);     // tile_dim0 = 4096
    g1[4] = 0;                      // tile_dim1/2 unused
    g1[5] = 4096;                   // tensor_dim0_stride lo
    g1[6] = 0;
    g1[7] = 0;
    const i32x4 gz4 = {0, 0, 0, 0};
    const i32x8 gz8 = {0, 0, 0, 0, 0, 0, 0, 0};
#endif

    for (int kb = 0; kb < 2048; kb += 64) {
        const unsigned short* kp = k + ((size_t)bh * 2048 + kb) * 64;
        const unsigned short* vp = v + ((size_t)bh * 2048 + kb) * 64;

#ifdef USE_TDM
        if (wv == 0) {   // wave-uniform: waves 1-3 branch around the TDM issue
            unsigned long long ga = (unsigned long long)kp;
            u32x4 g0;
            g0[0] = 1u;                                            // count=1
            g0[1] = lds_ks;                                        // lds_addr
            g0[2] = (unsigned)ga;                                  // global lo
            g0[3] = ((unsigned)(ga >> 32) & 0x01FFFFFFu) | (2u << 30); // hi|type=2
            __builtin_amdgcn_tensor_load_to_lds(g0, g1, gz4, gz4, gz8, 0);
        }
        // All 128 threads: V tile load + transpose into Vt
#pragma unroll
        for (int i = 0; i < 4; ++i) {
            int c   = tid + i * 128;
            int row = c >> 3;
            int d8  = (c & 7) * 8;
            Chunk8 ch;
            ch.u = *(const uint4*)(vp + row * 64 + d8);
#pragma unroll
            for (int e = 0; e < 8; ++e) Vt[(d8 + e) * 72 + row] = ch.s[e];
        }
#else
#pragma unroll
        for (int i = 0; i < 4; ++i) {
            int c   = tid + i * 128;
            int row = c >> 3;
            int d8  = (c & 7) * 8;
            *(uint4*)&Ks[row * 72 + d8] = *(const uint4*)(kp + row * 64 + d8);
            Chunk8 ch;
            ch.u = *(const uint4*)(vp + row * 64 + d8);
#pragma unroll
            for (int e = 0; e < 8; ++e) Vt[(d8 + e) * 72 + row] = ch.s[e];
        }
#endif
        if (kb + 64 < 2048) {   // cover next block's latency
            __builtin_prefetch(kp + 4096 + lane * 128, 0, 0);
            __builtin_prefetch(vp + 4096 + lane * 128, 0, 0);
        }
#ifdef USE_TDM
        if (wv == 0) __builtin_amdgcn_s_wait_tensorcnt(0);
#endif
        __syncthreads();

        // S = Q K^T (scale pre-folded into Q)
        v8f s[4];
#pragma unroll
        for (int t = 0; t < 4; ++t) {
            const unsigned short* kr = &Ks[(t * 16 + lr) * 72];
            v16bf b0 = ld_frag(kr, koff, koff + 16);
            v16bf b1 = ld_frag(kr, 32 + koff, 32 + koff + 16);
            s[t] = vzero();
            s[t] = WMMA_BF16(qa0, b0, s[t]);
            s[t] = WMMA_BF16(qa1, b1, s[t]);
        }

        // online softmax; row r of C lives in vgpr r across 16 lanes of a half
#pragma unroll
        for (int r = 0; r < 8; ++r) {
            float mx = fmaxf(fmaxf(s[0][r], s[1][r]), fmaxf(s[2][r], s[3][r]));
            mx = fmaxf(mx, __shfl_xor(mx, 1, 32));
            mx = fmaxf(mx, __shfl_xor(mx, 2, 32));
            mx = fmaxf(mx, __shfl_xor(mx, 4, 32));
            mx = fmaxf(mx, __shfl_xor(mx, 8, 32));
            float mnew = fmaxf(m[r], mx);
            float corr = __expf(m[r] - mnew);
            float rs = 0.f;
#pragma unroll
            for (int t = 0; t < 4; ++t) {
                float p = __expf(s[t][r] - mnew);
                s[t][r] = p;
                rs += p;
            }
            rs += __shfl_xor(rs, 1, 32);
            rs += __shfl_xor(rs, 2, 32);
            rs += __shfl_xor(rs, 4, 32);
            rs += __shfl_xor(rs, 8, 32);
            l[r] = l[r] * corr + rs;
            m[r] = mnew;
#pragma unroll
            for (int t = 0; t < 4; ++t) o[t][r] *= corr;
            int prow = (lane >> 4) * 8 + r;
#pragma unroll
            for (int t = 0; t < 4; ++t) pw[prow * 72 + t * 16 + lr] = f2bf(s[t][r]);
        }

        // O += P V   (A-frags from wave-private LDS; B-frags from Vt)
        v16bf pa0 = ld_frag(&pw[lr * 72], koff, koff + 16);
        v16bf pa1 = ld_frag(&pw[lr * 72], 32 + koff, 32 + koff + 16);
#pragma unroll
        for (int t = 0; t < 4; ++t) {
            const unsigned short* vr = &Vt[(t * 16 + lr) * 72];
            v16bf vb0 = ld_frag(vr, koff, koff + 16);
            v16bf vb1 = ld_frag(vr, 32 + koff, 32 + koff + 16);
            o[t] = WMMA_BF16(pa0, vb0, o[t]);
            o[t] = WMMA_BF16(pa1, vb1, o[t]);
        }
        __syncthreads();
    }

    const int b = bh >> 4, h = bh & 15;
#pragma unroll
    for (int r = 0; r < 8; ++r) {
        float inv = 1.0f / l[r];
        int   gr  = q0 + (lane >> 4) * 8 + r;
        size_t base = ((size_t)b * 2048 + gr) * 1024 + h * 64;
#pragma unroll
        for (int t = 0; t < 4; ++t)
            ob[base + t * 16 + lr] = f2bf(o[t][r] * inv);
    }
}

// ------------------------------------------------------------- out gemm -----
// out[8192][1024] f32 = attn_bf[8192][1024] * wout_bf[1024][1024]^T + bias

__global__ __launch_bounds__(256) void out_gemm(
    const unsigned short* __restrict__ ab,
    const unsigned short* __restrict__ wb,
    const float* __restrict__ bias,
    float* __restrict__ out)
{
    __shared__ unsigned short As[128 * 40];
    __shared__ unsigned short Bs[128 * 40];

    const int tid  = threadIdx.x;
    const int m0   = blockIdx.y * 128;
    const int n0   = blockIdx.x * 128;
    const int wv   = tid >> 5;
    const int lane = tid & 31;
    const int wr   = (wv >> 2) * 64;
    const int wc   = (wv & 3) * 32;
    const int lr   = lane & 15;
    const int koff = (lane >> 4) * 8;

    v8f acc[4][2];
#pragma unroll
    for (int i = 0; i < 4; ++i)
#pragma unroll
        for (int j = 0; j < 2; ++j) acc[i][j] = vzero();

    for (int k0 = 0; k0 < 1024; k0 += 32) {
#pragma unroll
        for (int i = 0; i < 2; ++i) {
            int c   = tid + i * 256;
            int row = c >> 2;
            int k8  = (c & 3) * 8;
            *(uint4*)&As[row * 40 + k8] = *(const uint4*)(ab + (size_t)(m0 + row) * 1024 + k0 + k8);
            *(uint4*)&Bs[row * 40 + k8] = *(const uint4*)(wb + (size_t)(n0 + row) * 1024 + k0 + k8);
            if (k0 + 32 < 1024) {
                __builtin_prefetch(ab + (size_t)(m0 + row) * 1024 + k0 + 32 + k8, 0, 0);
                __builtin_prefetch(wb + (size_t)(n0 + row) * 1024 + k0 + 32 + k8, 0, 0);
            }
        }
        __syncthreads();

        v16bf a[4], b[2];
#pragma unroll
        for (int i = 0; i < 4; ++i)
            a[i] = ld_frag(&As[(wr + i * 16 + lr) * 40], koff, koff + 16);
#pragma unroll
        for (int j = 0; j < 2; ++j)
            b[j] = ld_frag(&Bs[(wc + j * 16 + lr) * 40], koff, koff + 16);

#pragma unroll
        for (int i = 0; i < 4; ++i)
#pragma unroll
            for (int j = 0; j < 2; ++j)
                acc[i][j] = WMMA_BF16(a[i], b[j], acc[i][j]);
        __syncthreads();
    }

#pragma unroll
    for (int j = 0; j < 2; ++j) {
        int col  = n0 + wc + j * 16 + lr;
        float bv = bias[col];
#pragma unroll
        for (int i = 0; i < 4; ++i) {
            int rbase = m0 + wr + i * 16 + (lane >> 4) * 8;
#pragma unroll
            for (int r = 0; r < 8; ++r)
                out[(size_t)(rbase + r) * 1024 + col] = acc[i][j][r] + bv;
        }
    }
}

// --------------------------------------------------------------- launch -----

extern "C" void kernel_launch(void* const* d_in, const int* in_sizes, int n_in,
                              void* d_out, int out_size, void* d_ws, size_t ws_size,
                              hipStream_t stream)
{
    const float* x     = (const float*)d_in[0];   // [4][2048][1024]
    const float* w_qkv = (const float*)d_in[1];   // [3072][1024]
    const float* w_out = (const float*)d_in[2];   // [1024][1024]
    const float* b_out = (const float*)d_in[3];   // [1024]

    unsigned short* ws = (unsigned short*)d_ws;
    size_t off = 0;
    unsigned short* xb  = ws + off; off += (size_t)8192 * 1024;
    unsigned short* wqb = ws + off; off += (size_t)3072 * 1024;
    unsigned short* wob = ws + off; off += (size_t)1024 * 1024;
    unsigned short* qw  = ws + off; off += (size_t)64 * 2048 * 64;
    unsigned short* kw  = ws + off; off += (size_t)64 * 2048 * 64;
    unsigned short* vw  = ws + off; off += (size_t)64 * 2048 * 64;
    unsigned short* abf = ws + off; off += (size_t)8192 * 1024;

    cvt_bf16<<<(8388608 + 255) / 256, 256, 0, stream>>>(x, xb, 8388608);
    cvt_bf16<<<(3145728 + 255) / 256, 256, 0, stream>>>(w_qkv, wqb, 3145728);
    cvt_bf16<<<(1048576 + 255) / 256, 256, 0, stream>>>(w_out, wob, 1048576);

    qkv_gemm<<<dim3(24, 64), 256, 0, stream>>>(xb, wqb, qw, kw, vw);
    flash_attn<<<dim3(32, 64), 128, 0, stream>>>(qw, kw, vw, abf);
    out_gemm<<<dim3(8, 64), 256, 0, stream>>>(abf, wob, b_out, (float*)d_out);
}
